// LSTM_50749333569992
// MI455X (gfx1250) — matrix operand order repaired and checked by reading, not compiled, not run on hardware
//
#include <hip/hip_runtime.h>

// Problem sizes (match reference)
#define B_SZ   1024
#define T_SZ   512
#define D_SZ   64
#define H_SZ   32
#define G_SZ   128     // 4*H gates
#define OUT_SZ 2
#define EPS_BN 1e-5f

// Tiling
#define BLOCK_B 64                 // batch rows per workgroup
#define THREADS 256                // 8 waves: 4 m-tiles x 2 gate-halves
#define N_BLK   (B_SZ / BLOCK_B)   // 16 workgroups

// LDS strides (elements, even for dword-aligned bf16 pairs, padded vs banks)
#define XS  68
#define HS  36
#define WS1 66
#define WS2 34

// LDS layout (bytes)
#define OFF_XN   0                       // 64*68*2 = 8704 (pool head overlays this)
#define OFF_H1   8704                    // 64*36*2 = 4608
#define OFF_H2   13312                   // 4608
#define OFF_W1IH 17920                   // 128*66*2 = 16896
#define OFF_W1HH 34816                   // 128*34*2 = 8704
#define OFF_W2IH 43520                   // 8704
#define OFF_W2HH 52224                   // 8704
#define OFF_ZB1  60928                   // 128*4
#define OFF_ZB2  61440                   // 128*4
#define SMEM_SZ  61952

typedef __attribute__((ext_vector_type(16))) __bf16 v16bf;
typedef __attribute__((ext_vector_type(8)))  float  v8f;

union Frag16 { v16bf v; unsigned int u[8]; };

__device__ __forceinline__ unsigned short f2bf(float f) {
  unsigned int u = __float_as_uint(f);
  u += 0x7FFFu + ((u >> 16) & 1u);          // round-to-nearest-even
  return (unsigned short)(u >> 16);
}
__device__ __forceinline__ unsigned int pack2bf(float lo, float hi) {
  return (unsigned int)f2bf(lo) | ((unsigned int)f2bf(hi) << 16);
}
__device__ __forceinline__ float sigf(float z) { return 1.0f / (1.0f + __expf(-z)); }

// A fragment (16x32 bf16, MxK). base = &mat[row0*stride + col0].
// Lane L: M = L&15, K-base = (L>=16)?16:0; regs 0..3 -> K=kb+2v,2v+1; regs 4..7 -> K=8+kb+...
__device__ __forceinline__ v16bf load_a(const unsigned short* base, int stride, int lane) {
  const unsigned short* p = base + (lane & 15) * stride + ((lane >> 4) << 4);
  Frag16 f;
#pragma unroll
  for (int v = 0; v < 4; ++v) {
    f.u[v]     = *(const unsigned int*)(p + 2 * v);
    f.u[v + 4] = *(const unsigned int*)(p + 8 + 2 * v);
  }
  return f.v;
}

// B fragment (32x16 bf16, KxN) taken from row-major W[n][k] (z = x@W^T => B[k][n]=W[n][k]).
// Lane L: N = L&15, K-base = (L>=16)?16:0; reg v holds K = kb+2v, kb+2v+1.
__device__ __forceinline__ v16bf load_b(const unsigned short* base, int stride, int lane) {
  const unsigned short* p = base + (lane & 15) * stride + ((lane >> 4) << 4);
  Frag16 f;
#pragma unroll
  for (int v = 0; v < 8; ++v) f.u[v] = *(const unsigned int*)(p + 2 * v);
  return f.v;
}

__device__ __forceinline__ v8f wmma_bf16(v16bf a, v16bf b, v8f c) {
  return __builtin_amdgcn_wmma_f32_16x16x32_bf16(false, a, false, b, (short)0, c, false, false);
}

__global__ __launch_bounds__(THREADS)
void lstm2_bn_gap_fc_fused(const float* __restrict__ x,
                           const float* __restrict__ bn_gamma, const float* __restrict__ bn_beta,
                           const float* __restrict__ bn_mean,  const float* __restrict__ bn_var,
                           const float* __restrict__ Wih1, const float* __restrict__ Whh1,
                           const float* __restrict__ bih1, const float* __restrict__ bhh1,
                           const float* __restrict__ Wih2, const float* __restrict__ Whh2,
                           const float* __restrict__ bih2, const float* __restrict__ bhh2,
                           const float* __restrict__ fcW,  const float* __restrict__ fcb,
                           float* __restrict__ out) {
  __shared__ __align__(16) unsigned char smem[SMEM_SZ];
  unsigned short* s_xn   = (unsigned short*)(smem + OFF_XN);
  unsigned short* s_h1   = (unsigned short*)(smem + OFF_H1);
  unsigned short* s_h2   = (unsigned short*)(smem + OFF_H2);
  unsigned short* s_w1ih = (unsigned short*)(smem + OFF_W1IH);
  unsigned short* s_w1hh = (unsigned short*)(smem + OFF_W1HH);
  unsigned short* s_w2ih = (unsigned short*)(smem + OFF_W2IH);
  unsigned short* s_w2hh = (unsigned short*)(smem + OFF_W2HH);
  float* s_zb1  = (float*)(smem + OFF_ZB1);
  float* s_zb2  = (float*)(smem + OFF_ZB2);
  float* s_pool = (float*)(smem + OFF_XN);   // reused after the time loop

  const int tid  = threadIdx.x;
  const int lane = tid & 31;
  const int wave = tid >> 5;
  const int mt   = wave >> 1;                // m-tile (16 batch rows) 0..3
  const int half = wave & 1;                 // gate column half (h 0..15 or 16..31)
  const int b0   = blockIdx.x * BLOCK_B;

  // ---- stage weights/biases into LDS as bf16 (loop-invariant operands)
  for (int i = tid; i < G_SZ * D_SZ; i += THREADS) {
    int n = i >> 6, k = i & 63;
    s_w1ih[n * WS1 + k] = f2bf(Wih1[i]);
  }
  for (int i = tid; i < G_SZ * H_SZ; i += THREADS) {
    int n = i >> 5, k = i & 31;
    s_w1hh[n * WS2 + k] = f2bf(Whh1[i]);
    s_w2ih[n * WS2 + k] = f2bf(Wih2[i]);
    s_w2hh[n * WS2 + k] = f2bf(Whh2[i]);
  }
  for (int i = tid; i < G_SZ; i += THREADS) {
    s_zb1[i] = bih1[i] + bhh1[i];
    s_zb2[i] = bih2[i] + bhh2[i];
  }
  for (int i = tid; i < BLOCK_B * HS; i += THREADS) { s_h1[i] = 0; s_h2[i] = 0; }

  float c1[8], c2[8], pool[8];
#pragma unroll
  for (int r = 0; r < 8; ++r) { c1[r] = 0.f; c2[r] = 0.f; pool[r] = 0.f; }

  __syncthreads();

  const int xrow = tid >> 2;                 // 0..63
  const int xdp  = (tid & 3) << 4;           // 0,16,32,48
  const int hrow = mt * 16 + ((lane >> 4) << 3);   // D-layout: reg r -> M = r (+8 for hi lanes)
  const int hcol = half * 16 + (lane & 15);        // D-layout: lane -> N

  for (int t = 0; t < T_SZ; ++t) {
    // ---- BatchNorm over channel t (scalars uniform per step) + stage x tile as bf16
    float inv   = rsqrtf(bn_var[t] + EPS_BN);
    float scale = bn_gamma[t] * inv;
    float shift = bn_beta[t] - bn_mean[t] * scale;
    {
      const float* xr = x + ((size_t)(b0 + xrow) * T_SZ + t) * D_SZ + xdp;
      const float4* xp = (const float4*)xr;
      unsigned int* d32 = (unsigned int*)(s_xn + xrow * XS + xdp);
#pragma unroll
      for (int q = 0; q < 4; ++q) {
        float4 v = xp[q];
        d32[q * 2 + 0] = pack2bf(v.x * scale + shift, v.y * scale + shift);
        d32[q * 2 + 1] = pack2bf(v.z * scale + shift, v.w * scale + shift);
      }
      if (t + 1 < T_SZ) __builtin_prefetch(xr + D_SZ, 0, 1);  // next step's tile -> global_prefetch
    }
    __syncthreads();

    // ---- layer 1: z1 = xn@Wih1^T + h1@Whh1^T + (bih1+bhh1);  K = 32+32 (x) + 32 (h)
    v16bf ax0 = load_a(s_xn + mt * 16 * XS,      XS, lane);
    v16bf ax1 = load_a(s_xn + mt * 16 * XS + 32, XS, lane);
    v16bf ah1 = load_a(s_h1 + mt * 16 * HS,      HS, lane);
    v8f acc[4];
#pragma unroll
    for (int j = 0; j < 4; ++j) {            // j=0..3 -> gates i,f,g,o ; nt = 2j+half
      int nt = j * 2 + half;
      float bz = s_zb1[nt * 16 + (lane & 15)];
#pragma unroll
      for (int r = 0; r < 8; ++r) acc[j][r] = bz;
      acc[j] = wmma_bf16(ax0, load_b(s_w1ih + nt * 16 * WS1,      WS1, lane), acc[j]);
      acc[j] = wmma_bf16(ax1, load_b(s_w1ih + nt * 16 * WS1 + 32, WS1, lane), acc[j]);
      acc[j] = wmma_bf16(ah1, load_b(s_w1hh + nt * 16 * WS2,      WS2, lane), acc[j]);
    }
    float h1v[8];
#pragma unroll
    for (int r = 0; r < 8; ++r) {            // same lane/reg across j => same (b,h) element
      float ig = sigf(acc[0][r]);
      float fg = sigf(acc[1][r]);
      float gg = tanhf(acc[2][r]);
      float og = sigf(acc[3][r]);
      c1[r]  = fg * c1[r] + ig * gg;
      h1v[r] = og * tanhf(c1[r]);
    }
    __syncthreads();                          // pair-wave ah1 reads done before rewrite
#pragma unroll
    for (int r = 0; r < 8; ++r) s_h1[(hrow + r) * HS + hcol] = f2bf(h1v[r]);
    __syncthreads();

    // ---- layer 2: z2 = h1_t@Wih2^T + h2@Whh2^T + (bih2+bhh2);  K = 32 + 32
    v16bf a1 = load_a(s_h1 + mt * 16 * HS, HS, lane);
    v16bf a2 = load_a(s_h2 + mt * 16 * HS, HS, lane);
#pragma unroll
    for (int j = 0; j < 4; ++j) {
      int nt = j * 2 + half;
      float bz = s_zb2[nt * 16 + (lane & 15)];
#pragma unroll
      for (int r = 0; r < 8; ++r) acc[j][r] = bz;
      acc[j] = wmma_bf16(a1, load_b(s_w2ih + nt * 16 * WS2, WS2, lane), acc[j]);
      acc[j] = wmma_bf16(a2, load_b(s_w2hh + nt * 16 * WS2, WS2, lane), acc[j]);
    }
    float h2v[8];
#pragma unroll
    for (int r = 0; r < 8; ++r) {
      float ig = sigf(acc[0][r]);
      float fg = sigf(acc[1][r]);
      float gg = tanhf(acc[2][r]);
      float og = sigf(acc[3][r]);
      c2[r]   = fg * c2[r] + ig * gg;
      h2v[r]  = og * tanhf(c2[r]);
      pool[r] += h2v[r];                      // GAP accumulates in registers
    }
    __syncthreads();                          // pair-wave a2 reads done before rewrite
#pragma unroll
    for (int r = 0; r < 8; ++r) s_h2[(hrow + r) * HS + hcol] = f2bf(h2v[r]);
    __syncthreads();
  }

  // ---- GAP scale + FC head (pool overlays the now-dead x tile)
#pragma unroll
  for (int r = 0; r < 8; ++r) s_pool[(hrow + r) * 33 + hcol] = pool[r] * (1.0f / (float)T_SZ);
  __syncthreads();
  if (tid < BLOCK_B * OUT_SZ) {
    int bl = tid >> 1;
    int o  = tid & 1;
    float sum = fcb[o];
#pragma unroll
    for (int h = 0; h < H_SZ; ++h) sum += s_pool[bl * 33 + h] * fcW[o * H_SZ + h];
    out[(size_t)(b0 + bl) * OUT_SZ + o] = sum;
  }
}

extern "C" void kernel_launch(void* const* d_in, const int* in_sizes, int n_in,
                              void* d_out, int out_size, void* d_ws, size_t ws_size,
                              hipStream_t stream) {
  (void)in_sizes; (void)n_in; (void)out_size; (void)d_ws; (void)ws_size;
  const float* x        = (const float*)d_in[0];
  const float* bn_gamma = (const float*)d_in[1];
  const float* bn_beta  = (const float*)d_in[2];
  const float* bn_mean  = (const float*)d_in[3];
  const float* bn_var   = (const float*)d_in[4];
  const float* Wih1     = (const float*)d_in[5];
  const float* Whh1     = (const float*)d_in[6];
  const float* bih1     = (const float*)d_in[7];
  const float* bhh1     = (const float*)d_in[8];
  const float* Wih2     = (const float*)d_in[9];
  const float* Whh2     = (const float*)d_in[10];
  const float* bih2     = (const float*)d_in[11];
  const float* bhh2     = (const float*)d_in[12];
  const float* fcW      = (const float*)d_in[13];
  const float* fcb      = (const float*)d_in[14];
  float* out            = (float*)d_out;

  lstm2_bn_gap_fc_fused<<<dim3(N_BLK), dim3(THREADS), 0, stream>>>(
      x, bn_gamma, bn_beta, bn_mean, bn_var,
      Wih1, Whh1, bih1, bhh1, Wih2, Whh2, bih2, bhh2, fcW, fcb, out);
}
